// LocalConvNet_90185723281824
// MI455X (gfx1250) — matrix operand order repaired and checked by reading, compiled, not run on hardware
//
#include <hip/hip_runtime.h>
#include <hip/hip_bf16.h>
#include <cstddef>

typedef _Float16 v16h __attribute__((ext_vector_type(16)));
typedef _Float16 v8h  __attribute__((ext_vector_type(8)));
typedef float    v8f  __attribute__((ext_vector_type(8)));

#define NPTS 4096
#define EPS  1e-5f

// ---------------------------------------------------------------------------
// WMMA helpers (CDNA5 wave32, v_wmma_f32_16x16x32_f16)
// ---------------------------------------------------------------------------
__device__ __forceinline__ v8f vzero8() {
    v8f c;
#pragma unroll
    for (int i = 0; i < 8; ++i) c[i] = 0.0f;
    return c;
}

__device__ __forceinline__ v8f wmma_step(v16h a, v16h b, v8f c) {
    return __builtin_amdgcn_wmma_f32_16x16x32_f16(
        /*neg_a=*/false, a, /*neg_b=*/false, b,
        /*c_mod=*/(short)0, c, /*reuse_a=*/false, /*reuse_b=*/false);
}

// A fragment (16x32 f16): lane m=L%16 holds K = h*16+{0..15}.
// stride==0 gives a row-broadcast A tile (all 16 rows identical).
__device__ __forceinline__ v16h load_a(const _Float16* base, int stride) {
    int lane = threadIdx.x & 31;
    int m = lane & 15, h = lane >> 4;
    const _Float16* p = base + m * stride + h * 16;
    v8h lo = *(const v8h*)(p);
    v8h hi = *(const v8h*)(p + 8);
    v16h a;
#pragma unroll
    for (int i = 0; i < 8; ++i) { a[i] = lo[i]; a[i + 8] = hi[i]; }
    return a;
}

// B fragment (32x16 f16): lane n=L%16 holds K = h*16+{0..15} of column n.
// With weights stored transposed row-major Wt[cout][cin], this is one
// contiguous 32-byte load from the output-channel row.
__device__ __forceinline__ v16h load_b(const _Float16* p) {
    return *(const v16h*)(p);
}

// ---------------------------------------------------------------------------
// Weight prep: fold BN into transposed f16 weights + f32 bias
//   Wt[o][c] = W[c][o] * gamma[o]*rsqrt(var[o]+eps)
//   bias[o]  = (b[o]-mean[o])*scale + beta[o]
// ---------------------------------------------------------------------------
__global__ void prep_weights_kernel(const float* __restrict__ W,
                                    const float* __restrict__ bv,
                                    const float* __restrict__ gamma,
                                    const float* __restrict__ beta,
                                    const float* __restrict__ mean,
                                    const float* __restrict__ var,
                                    int cin, int cout,
                                    _Float16* __restrict__ wt,
                                    float* __restrict__ bias) {
    int o = blockIdx.x;
    float s = gamma[o] * rsqrtf(var[o] + EPS);
    for (int c = threadIdx.x; c < cin; c += blockDim.x)
        wt[o * cin + c] = (_Float16)(W[c * cout + o] * s);
    if (threadIdx.x == 0)
        bias[o] = (bv[o] - mean[o]) * s + beta[o];
}

// ---------------------------------------------------------------------------
// KNN: 256 queries per block, candidates staged through LDS, register-resident
// sorted top-16 (fully unrolled insertion -> cndmask, no scratch spills).
// ---------------------------------------------------------------------------
__global__ __launch_bounds__(256) void knn_kernel(const float* __restrict__ pos,
                                                  int* __restrict__ idxout) {
    __shared__ float sx[256], sy[256], sz[256];
    int b   = blockIdx.x >> 4;
    int n0  = (blockIdx.x & 15) << 8;
    int tid = threadIdx.x;
    int q   = n0 + tid;
    const float* pb = pos + (size_t)b * NPTS * 3;
    float qx = pb[q * 3 + 0], qy = pb[q * 3 + 1], qz = pb[q * 3 + 2];

    float bd[16]; int bi[16];
#pragma unroll
    for (int i = 0; i < 16; ++i) { bd[i] = 3.4e38f; bi[i] = 0; }

    for (int c0 = 0; c0 < NPTS; c0 += 256) {
        __syncthreads();
        sx[tid] = pb[(c0 + tid) * 3 + 0];
        sy[tid] = pb[(c0 + tid) * 3 + 1];
        sz[tid] = pb[(c0 + tid) * 3 + 2];
        __syncthreads();
        for (int j = 0; j < 256; ++j) {
            float dx = qx - sx[j], dy = qy - sy[j], dz = qz - sz[j];
            float d = dx * dx + dy * dy + dz * dz;
            int cand = c0 + j;
            if (d < bd[15]) {
#pragma unroll
                for (int s = 15; s >= 0; --s) {
                    float pd = (s == 0) ? -3.4e38f : bd[s - 1];
                    int   pi = (s == 0) ? cand     : bi[s - 1];
                    if (d < bd[s]) {
                        bool shift = (d < pd);
                        bd[s] = shift ? pd : d;
                        bi[s] = shift ? pi : cand;
                    }
                }
            }
        }
    }
    int* op = idxout + ((size_t)b * NPTS + q) * 16;
#pragma unroll
    for (int i = 0; i < 16; ++i) op[i] = bi[i];
}

// ---------------------------------------------------------------------------
// Generic WMMA layer tile: one wave computes rows of one 16-row M-tile
// (Atile already offset to the wave's rows), looping its assigned N-tiles.
// ---------------------------------------------------------------------------
__device__ __forceinline__ void mlp_tile(const _Float16* Atile, int astride, int Ktot,
                                         const _Float16* W, int cin,
                                         _Float16* O, int ostride,
                                         const float* bias,
                                         int nt0, int nt1, int ntstep, bool relu) {
    int lane = threadIdx.x & 31;
    int n = lane & 15, h = lane >> 4;
    for (int nt = nt0; nt < nt1; nt += ntstep) {
        v8f c = vzero8();
        for (int k0 = 0; k0 < Ktot; k0 += 32) {
            v16h a = load_a(Atile + k0, astride);
            v16h bb = load_b(W + (size_t)(nt * 16 + n) * cin + k0 + h * 16);
            c = wmma_step(a, bb, c);
        }
        float bvv = bias[nt * 16 + n];
#pragma unroll
        for (int r = 0; r < 8; ++r) {
            float v = c[r] + bvv;
            if (relu) v = fmaxf(v, 0.0f);
            O[(size_t)(r + h * 8) * ostride + nt * 16 + n] = (_Float16)v;
        }
    }
}

// ---------------------------------------------------------------------------
// Fused pipeline: workgroup = 8 points (128 gathered rows, 8 M-tiles, 8 waves)
// LDS (64KB) lifetime-overlapped regions.
// ---------------------------------------------------------------------------
__global__ __launch_bounds__(256) void fused_kernel(const float* __restrict__ x,
                                                    const int* __restrict__ knn,
                                                    const _Float16* __restrict__ wt,
                                                    const float* __restrict__ bias,
                                                    float* __restrict__ out) {
    __shared__ __attribute__((aligned(64))) _Float16 smem[32768];  // 64 KB

    _Float16* X     = smem;            // 128x64  gathered input / h2
    _Float16* H1    = smem + 8192;     // 128x64
    _Float16* G     = smem + 16384;    // 128x128 local-MLP output
    _Float16* GMAX  = smem + 8192;     // 16x128  (over H1, dead)
    _Float16* GMID  = smem + 10240;    // 16x256
    _Float16* GLB   = smem + 14336;    // 16x128
    _Float16* C1B   = smem;            // 128x128 (over X+H1, written post-barrier)
    _Float16* CMAX  = smem + 16384;    // 16x128  (over G, dead)
    _Float16* CGMID = smem + 18432;    // 16x256

    const _Float16* wL1  = wt;
    const _Float16* wL2  = wt + 4096;
    const _Float16* wL3  = wt + 8192;
    const _Float16* wG1  = wt + 16384;
    const _Float16* wG2  = wt + 49152;
    const _Float16* wC1  = wt + 81920;
    const _Float16* wC2  = wt + 114688;
    const _Float16* wCG1 = wt + 131072;
    const _Float16* wCG2 = wt + 163840;

    const float* bL1  = bias;
    const float* bL2  = bias + 64;
    const float* bL3  = bias + 128;
    const float* bG1  = bias + 256;
    const float* bG2  = bias + 512;
    const float* bC1  = bias + 640;
    const float* bC2  = bias + 768;
    const float* bCG1 = bias + 896;
    const float* bCG2 = bias + 1152;

    int tid  = threadIdx.x;
    int wave = tid >> 5;
    int lane = tid & 31;
    int n = lane & 15, h = lane >> 4;
    int b  = blockIdx.x >> 9;        // 512 tiles per batch
    int n0 = (blockIdx.x & 511) << 3;

    // --- gather neighbor features -> f16 in LDS (128 rows x 64 ch) ---------
    {
        int r  = tid >> 1;
        int c0 = (tid & 1) << 5;
        int p = r >> 4, k = r & 15;
        int src = knn[((size_t)b * NPTS + n0 + p) * 16 + k];
        const float* xr = x + ((size_t)b * NPTS + src) * 64 + c0;
#pragma unroll
        for (int j = 0; j < 32; ++j)
            X[(size_t)r * 64 + c0 + j] = (_Float16)xr[j];
    }
    __syncthreads();

    // --- local MLP: 64 -> 64 -> 64 -> 128 ----------------------------------
    mlp_tile(X + wave * 16 * 64, 64, 64, wL1, 64, H1 + wave * 16 * 64, 64, bL1, 0, 4, 1, true);
    __syncthreads();
    mlp_tile(H1 + wave * 16 * 64, 64, 64, wL2, 64, X + wave * 16 * 64, 64, bL2, 0, 4, 1, true);
    __syncthreads();
    mlp_tile(X + wave * 16 * 64, 64, 64, wL3, 64, G + wave * 16 * 128, 128, bL3, 0, 8, 1, false);
    __syncthreads();

    // --- max over K (16 neighbor rows per point), pad rows 8..15 with 0 ----
    for (int i = tid; i < 2048; i += 256) {
        int p = i >> 7, c = i & 127;
        float m = 0.0f;
        if (p < 8) {
            m = -3.4e38f;
            for (int k = 0; k < 16; ++k)
                m = fmaxf(m, (float)G[(size_t)(p * 16 + k) * 128 + c]);
        }
        GMAX[i] = (_Float16)m;
    }
    __syncthreads();

    // --- glb MLP: 128 -> 256 (relu) -> 128 ---------------------------------
    mlp_tile(GMAX, 128, 128, wG1, 128, GMID, 256, bG1, wave, 16, 8, true);
    __syncthreads();
    mlp_tile(GMID, 256, 256, wG2, 256, GLB, 128, bG2, wave, wave + 1, 1, false);
    __syncthreads();

    // --- comb layer 1: concat(g,glb)=K256 -> 128, relu ---------------------
    // K-steps 0..3 read g; 4..7 read broadcast glb row (stride-0 A tile).
    {
        v8f acc[8];
#pragma unroll
        for (int nt = 0; nt < 8; ++nt) acc[nt] = vzero8();
        const _Float16* Ag  = G + wave * 16 * 128;
        const _Float16* Agb = GLB + wave * 128;
        for (int k0 = 0; k0 < 256; k0 += 32) {
            v16h a = (k0 < 128) ? load_a(Ag + k0, 128) : load_a(Agb + (k0 - 128), 0);
#pragma unroll
            for (int nt = 0; nt < 8; ++nt) {
                v16h bb = load_b(wC1 + (size_t)(nt * 16 + n) * 256 + k0 + h * 16);
                acc[nt] = wmma_step(a, bb, acc[nt]);
            }
        }
        __syncthreads();   // G and GLB dead; safe to overwrite smem[0..16384)
#pragma unroll
        for (int nt = 0; nt < 8; ++nt) {
            float bvv = bC1[nt * 16 + n];
#pragma unroll
            for (int r = 0; r < 8; ++r) {
                float v = fmaxf(acc[nt][r] + bvv, 0.0f);
                C1B[(size_t)(wave * 16 + r + h * 8) * 128 + nt * 16 + n] = (_Float16)v;
            }
        }
    }
    __syncthreads();

    // --- comb layer 2: 128 -> 128 (no relu) fused with max over K ----------
    for (int nt = 0; nt < 8; ++nt) {
        v8f c = vzero8();
        for (int k0 = 0; k0 < 128; k0 += 32) {
            v16h a = load_a(C1B + wave * 16 * 128 + k0, 128);
            v16h bb = load_b(wC2 + (size_t)(nt * 16 + n) * 128 + k0 + h * 16);
            c = wmma_step(a, bb, c);
        }
        float bvv = bC2[nt * 16 + n];
        float m = -3.4e38f;
#pragma unroll
        for (int r = 0; r < 8; ++r) m = fmaxf(m, c[r] + bvv);
        m = fmaxf(m, __shfl_xor(m, 16, 32));       // combine both M-halves
        if (h == 0) CMAX[wave * 128 + nt * 16 + n] = (_Float16)m;
    }
    for (int i = tid; i < 1024; i += 256)          // zero pad rows 8..15
        CMAX[1024 + i] = (_Float16)0.0f;
    __syncthreads();

    // --- comb_glb MLP: 128 -> 256 (relu) -> 128, write to global -----------
    mlp_tile(CMAX, 128, 128, wCG1, 128, CGMID, 256, bCG1, wave, 16, 8, true);
    __syncthreads();
    {
        int nt = wave;
        v8f c = vzero8();
        for (int k0 = 0; k0 < 256; k0 += 32) {
            v16h a = load_a(CGMID + k0, 256);
            v16h bb = load_b(wCG2 + (size_t)(nt * 16 + n) * 256 + k0 + h * 16);
            c = wmma_step(a, bb, c);
        }
        float bvv = bCG2[nt * 16 + n];
        if (h == 0) {   // M = r (points 0..7) live in lanes 0..15
#pragma unroll
            for (int r = 0; r < 8; ++r)
                out[((size_t)b * NPTS + n0 + r) * 128 + nt * 16 + n] = c[r] + bvv;
        }
    }
}

// ---------------------------------------------------------------------------
// Host launcher
// ---------------------------------------------------------------------------
extern "C" void kernel_launch(void* const* d_in, const int* in_sizes, int n_in,
                              void* d_out, int out_size, void* d_ws, size_t ws_size,
                              hipStream_t stream) {
    (void)in_sizes; (void)n_in; (void)out_size; (void)ws_size;
    const float* x   = (const float*)d_in[0];
    const float* pos = (const float*)d_in[1];

    char* ws = (char*)d_ws;
    int*       idx   = (int*)ws;                                // 1 MB
    _Float16*  wtb   = (_Float16*)(ws + (1 << 20));             // 384 KB f16
    float*     bb    = (float*)(ws + (1 << 20) + 393216);       // 5 KB f32

    _Float16* wL1  = wtb;            float* bL1  = bb;
    _Float16* wL2  = wtb + 4096;     float* bL2  = bb + 64;
    _Float16* wL3  = wtb + 8192;     float* bL3  = bb + 128;
    _Float16* wG1  = wtb + 16384;    float* bG1  = bb + 256;
    _Float16* wG2  = wtb + 49152;    float* bG2  = bb + 512;
    _Float16* wC1  = wtb + 81920;    float* bC1  = bb + 640;
    _Float16* wC2  = wtb + 114688;   float* bC2  = bb + 768;
    _Float16* wCG1 = wtb + 131072;   float* bCG1 = bb + 896;
    _Float16* wCG2 = wtb + 163840;   float* bCG2 = bb + 1152;

    // d_in layout: x, pos, then 9 layers x (W,b,gamma,beta,mean,var)
    #define PREP(base, cin, cout, wt, bs)                                        \
        prep_weights_kernel<<<(cout), 64, 0, stream>>>(                          \
            (const float*)d_in[(base)+0], (const float*)d_in[(base)+1],          \
            (const float*)d_in[(base)+2], (const float*)d_in[(base)+3],          \
            (const float*)d_in[(base)+4], (const float*)d_in[(base)+5],          \
            (cin), (cout), (wt), (bs))

    PREP(2,  64,  64,  wL1,  bL1);   // local layer 0
    PREP(8,  64,  64,  wL2,  bL2);   // local layer 1
    PREP(14, 64,  128, wL3,  bL3);   // local layer 2
    PREP(20, 128, 256, wG1,  bG1);   // glb layer 0
    PREP(26, 256, 128, wG2,  bG2);   // glb layer 1
    PREP(32, 256, 128, wC1,  bC1);   // comb layer 0
    PREP(38, 128, 128, wC2,  bC2);   // comb layer 1
    PREP(44, 128, 256, wCG1, bCG1);  // comb_glb layer 0
    PREP(50, 256, 128, wCG2, bCG2);  // comb_glb layer 1
    #undef PREP

    knn_kernel<<<64, 256, 0, stream>>>(pos, idx);

    // 4 batches * 512 point-tiles of 8 points
    fused_kernel<<<2048, 256, 0, stream>>>(x, idx, wtb, bb, (float*)d_out);
}
